// Transformer_49615462204145
// MI455X (gfx1250) — compile-verified, matrix operand order checked by reading, not run on hardware
//
#include <hip/hip_runtime.h>

// ---------------------------------------------------------------------------
// MI455X (gfx1250, wave32) transformer forward.
//  * all matrix math: v_wmma_f32_16x16x32_bf16 (fp32 accumulate)
//  * all hot-loop staging: global_load_async_to_lds_b128 + s_wait_asynccnt
//    (ASYNCcnt path, double-buffered LDS in the GEMM)
//  * fp32->bf16 conversion done once per tensor in cheap pre-passes; weights
//    converted+transposed to [N][K] so B tiles are contiguous along K.
// Shapes: D=1024 H=16 HD=64 F=4096 V=32000 L=2048 LAT=512 WIN=512 NL=4
// ---------------------------------------------------------------------------

typedef __attribute__((ext_vector_type(16))) __bf16 bf16x16;
typedef __attribute__((ext_vector_type(8)))  __bf16 bf16x8;
typedef __attribute__((ext_vector_type(4)))  __bf16 bf16x4;
typedef __attribute__((ext_vector_type(8)))  float  f32x8;

#define LSEQ 2048
#define DMODEL 1024
#define NHEAD 16
#define HDIM 64
#define FDIM 4096
#define VOCAB 32000
#define LATD 512

__device__ __forceinline__ f32x8 zero8() {
  f32x8 v;
#pragma unroll
  for (int i = 0; i < 8; ++i) v[i] = 0.0f;
  return v;
}

// A/B fragment from LDS row [row][k] (16B-aligned rows).
// element i -> k = (i>>3)*16 + half*8 + (i&7)
__device__ __forceinline__ bf16x16 load_frag(const __bf16* rowp, int half) {
  bf16x8 lo = *(const bf16x8*)(rowp + half * 8);
  bf16x8 hi = *(const bf16x8*)(rowp + 16 + half * 8);
  bf16x16 r;
#pragma unroll
  for (int i = 0; i < 8; ++i) { r[i] = lo[i]; r[i + 8] = hi[i]; }
  return r;
}

__device__ __forceinline__ f32x8 wmma_bf16(bf16x16 a, bf16x16 b, f32x8 c) {
  return __builtin_amdgcn_wmma_f32_16x16x32_bf16(false, a, false, b,
                                                 (short)0, c, false, false);
}

// async 16B copy: global (bf16) -> LDS, tracked by ASYNCcnt
__device__ __forceinline__ void async_cp16(const __bf16* gp, const __bf16* lp) {
  const unsigned long long ga = (unsigned long long)(uintptr_t)gp;
  const unsigned lo = (unsigned)(uintptr_t)lp;
  asm volatile("global_load_async_to_lds_b128 %0, %1, off"
               :: "v"(lo), "v"(ga) : "memory");
}

// ---------------------------------------------------------------------------
// GEMM: C[M,N] = A[M,K] (bf16) * B[N,K] (bf16, K-contiguous) [+C]
// Block = 128 threads (4 waves). Block tile 128x64, K-tile 32.
// Wave w owns 32x64 (2x4 WMMA frags -> 8 v_wmma per K-step).
// Double-buffered LDS, staged entirely with async-to-LDS DMA.
// CMODE: 0 = fp32 store, 1 = fp32 accumulate, 2 = bf16 store,
//        3 = bf16 transposed store (Ct[N][M], one b128 store per frag)
// ---------------------------------------------------------------------------
template <int CMODE>
__global__ void __launch_bounds__(128)
gemm_bf16(const __bf16* __restrict__ A, const __bf16* __restrict__ B,
          void* __restrict__ Cv, int M, int N, int K) {
  __shared__ __align__(16) __bf16 As[2][128][40];  // row stride 80B = 16*5
  __shared__ __align__(16) __bf16 Bs[2][64][40];

  const int tid  = threadIdx.x;
  const int w    = tid >> 5, lane = tid & 31;
  const int half = lane >> 4, lm = lane & 15;
  const int m0   = blockIdx.y * 128, n0 = blockIdx.x * 64;

  auto issue_tile = [&](int kt, int buf) {
    const int k0 = kt << 5;
#pragma unroll
    for (int i = 0; i < 4; ++i) {          // A: 128 rows x 4 chunks of 16B
      const int ca = tid + i * 128;
      const int r = ca >> 2, c = ca & 3;
      async_cp16(A + (size_t)(m0 + r) * K + k0 + c * 8, &As[buf][r][c * 8]);
    }
#pragma unroll
    for (int i = 0; i < 2; ++i) {          // B: 64 rows x 4 chunks of 16B
      const int cb = tid + i * 128;
      const int r = cb >> 2, c = cb & 3;
      async_cp16(B + (size_t)(n0 + r) * K + k0 + c * 8, &Bs[buf][r][c * 8]);
    }
  };

  f32x8 acc[2][4];
#pragma unroll
  for (int i = 0; i < 2; ++i)
#pragma unroll
    for (int j = 0; j < 4; ++j) acc[i][j] = zero8();

  const int ktiles = K >> 5;
  issue_tile(0, 0);
  if (ktiles > 1) issue_tile(1, 1);

  for (int kt = 0; kt < ktiles; ++kt) {
    const int buf = kt & 1;
    if (kt + 1 < ktiles)   // next tile (6 ops/wave) may stay in flight
      asm volatile("s_wait_asynccnt 6" ::: "memory");
    else
      asm volatile("s_wait_asynccnt 0" ::: "memory");
    __syncthreads();

    bf16x16 af[2], bfr[4];
#pragma unroll
    for (int fm = 0; fm < 2; ++fm)
      af[fm] = load_frag(&As[buf][w * 32 + fm * 16 + lm][0], half);
#pragma unroll
    for (int fn = 0; fn < 4; ++fn)
      bfr[fn] = load_frag(&Bs[buf][fn * 16 + lm][0], half);
#pragma unroll
    for (int fm = 0; fm < 2; ++fm)
#pragma unroll
      for (int fn = 0; fn < 4; ++fn)
        acc[fm][fn] = wmma_bf16(af[fm], bfr[fn], acc[fm][fn]);

    __syncthreads();                       // all waves done with buf
    if (kt + 2 < ktiles) issue_tile(kt + 2, buf);
  }

  // C frag layout: element r of v8f -> m = r + 8*half, n = lane&15
#pragma unroll
  for (int fm = 0; fm < 2; ++fm)
#pragma unroll
    for (int fn = 0; fn < 4; ++fn) {
      const int n = n0 + fn * 16 + lm;
      if (CMODE == 3) {                    // transposed bf16: one 16B store
        __bf16* Ct = (__bf16*)Cv;
        const int mb = m0 + w * 32 + fm * 16 + 8 * half;
        bf16x8 v;
#pragma unroll
        for (int r = 0; r < 8; ++r) v[r] = (__bf16)acc[fm][fn][r];
        *(bf16x8*)(Ct + (size_t)n * M + mb) = v;
      } else {
#pragma unroll
        for (int r = 0; r < 8; ++r) {
          const int m = m0 + w * 32 + fm * 16 + r + 8 * half;
          const float v = acc[fm][fn][r];
          if (CMODE == 0)      ((float*)Cv)[(size_t)m * N + n] = v;
          else if (CMODE == 1) ((float*)Cv)[(size_t)m * N + n] += v;
          else                 ((__bf16*)Cv)[(size_t)m * N + n] = (__bf16)v;
        }
      }
    }
}

// ---------------------------------------------------------------------------
// Flash attention, all-bf16 inputs (Q,K row-major [l][1024]; V pre-transposed
// [1024][L]). K/V tiles staged with async-to-LDS DMA. fp32 online softmax.
// Grid (L/64, H), 128 threads; wave w owns query rows qb*64+w*16..+15.
// ---------------------------------------------------------------------------
template <int IS_SWA>
__global__ void __launch_bounds__(128)
flash_attn(const __bf16* __restrict__ Q, const __bf16* __restrict__ Kg,
           const __bf16* __restrict__ Vt, __bf16* __restrict__ O) {
  __shared__ __align__(16) __bf16 Ks[64][72];      // [key][hd]
  __shared__ __align__(16) __bf16 Vs[64][72];      // [hd][key]
  __shared__ __align__(16) __bf16 Ps[4][16][72];   // per-wave P [m][key]

  const int tid  = threadIdx.x;
  const int w    = tid >> 5, lane = tid & 31;
  const int half = lane >> 4, lm = lane & 15;
  const int qb   = blockIdx.x, h = blockIdx.y;

  // Q fragments straight from global bf16 (two 16B loads per frag)
  bf16x16 qa[2];
  {
    const __bf16* qrow = Q + (size_t)(qb * 64 + w * 16 + lm) * DMODEL + h * HDIM;
#pragma unroll
    for (int ks = 0; ks < 2; ++ks)
      qa[ks] = load_frag(qrow + ks * 32, half);
  }

  f32x8 o[4];
#pragma unroll
  for (int i = 0; i < 4; ++i) o[i] = zero8();
  float mrow[8], lrow[8];
#pragma unroll
  for (int r = 0; r < 8; ++r) { mrow[r] = -1e30f; lrow[r] = 0.0f; }

  const int kb_lo = IS_SWA ? ((qb > 8) ? qb - 8 : 0) : 0;
  for (int kb = kb_lo; kb <= qb; ++kb) {
    __syncthreads();                       // previous tile fully consumed
#pragma unroll
    for (int i = 0; i < 4; ++i) {          // 64 rows x 8 chunks of 16B each
      const int c = tid + i * 128;
      const int r = c >> 3, cc = c & 7;
      async_cp16(Kg + (size_t)(kb * 64 + r) * DMODEL + h * HDIM + cc * 8,
                 &Ks[r][cc * 8]);
      async_cp16(Vt + (size_t)(h * HDIM + r) * LSEQ + kb * 64 + cc * 8,
                 &Vs[r][cc * 8]);
    }
    asm volatile("s_wait_asynccnt 0" ::: "memory");
    __syncthreads();

    // S = Q K^T  (16 q-rows x 64 keys per wave)
    f32x8 s[4];
#pragma unroll
    for (int fn = 0; fn < 4; ++fn) s[fn] = zero8();
#pragma unroll
    for (int ks = 0; ks < 2; ++ks)
#pragma unroll
      for (int fn = 0; fn < 4; ++fn) {
        const bf16x16 bk = load_frag(&Ks[fn * 16 + lm][ks * 32], half);
        s[fn] = wmma_bf16(qa[ks], bk, s[fn]);
      }

    // scale, mask, online softmax (rows m = r + 8*half)
#pragma unroll
    for (int r = 0; r < 8; ++r) {
      const int q_idx = qb * 64 + w * 16 + r + 8 * half;
      float mx = -1e30f;
#pragma unroll
      for (int fn = 0; fn < 4; ++fn) {
        const int k_idx = kb * 64 + fn * 16 + lm;
        float v = s[fn][r] * 0.125f;
        const bool ok = (k_idx <= q_idx) && (!IS_SWA || (k_idx > q_idx - 512));
        v = ok ? v : -1e30f;
        s[fn][r] = v;
        mx = fmaxf(mx, v);
      }
#pragma unroll
      for (int off = 1; off < 16; off <<= 1)
        mx = fmaxf(mx, __shfl_xor(mx, off, 32));
      const float mnew = fmaxf(mrow[r], mx);
      const float corr = __expf(mrow[r] - mnew);
      float psum = 0.0f;
#pragma unroll
      for (int fn = 0; fn < 4; ++fn) {
        const float p = __expf(s[fn][r] - mnew);
        psum += p;
        Ps[w][r + 8 * half][fn * 16 + lm] = (__bf16)p;
      }
#pragma unroll
      for (int off = 1; off < 16; off <<= 1)
        psum += __shfl_xor(psum, off, 32);
      lrow[r] = lrow[r] * corr + psum;
      mrow[r] = mnew;
#pragma unroll
      for (int fo = 0; fo < 4; ++fo) o[fo][r] *= corr;
    }

    // O += P V
#pragma unroll
    for (int ks = 0; ks < 2; ++ks) {
      const bf16x16 pa = load_frag(&Ps[w][lm][ks * 32], half);
#pragma unroll
      for (int fo = 0; fo < 4; ++fo) {
        const bf16x16 vb = load_frag(&Vs[fo * 16 + lm][ks * 32], half);
        o[fo] = wmma_bf16(pa, vb, o[fo]);
      }
    }
  }

#pragma unroll
  for (int fo = 0; fo < 4; ++fo)
#pragma unroll
    for (int r = 0; r < 8; ++r) {
      const int m = qb * 64 + w * 16 + r + 8 * half;
      O[(size_t)m * DMODEL + h * HDIM + fo * 16 + lm] =
          (__bf16)(o[fo][r] / lrow[r]);
    }
}

// ---------------------------------------------------------------------------
// Conversion / elementwise kernels
// ---------------------------------------------------------------------------
// fp32 [K][N] -> bf16 [N][K] (tiled transpose through LDS)
__global__ void __launch_bounds__(256)
convtr_k(const float* __restrict__ W, __bf16* __restrict__ Wt, int K, int N) {
  __shared__ float T[64][65];
  const int k0 = blockIdx.y * 64, n0 = blockIdx.x * 64;
  const int tid = threadIdx.x;
#pragma unroll
  for (int i = 0; i < 4; ++i) {
    const int idx = tid + i * 256;         // 64 rows x 16 float4
    const int r = idx >> 4, c4 = idx & 15;
    const float4 f = *(const float4*)(W + (size_t)(k0 + r) * N + n0 + c4 * 4);
    T[r][c4 * 4 + 0] = f.x; T[r][c4 * 4 + 1] = f.y;
    T[r][c4 * 4 + 2] = f.z; T[r][c4 * 4 + 3] = f.w;
  }
  __syncthreads();
#pragma unroll
  for (int i = 0; i < 4; ++i) {
    const int idx = tid + i * 256;         // 64 n-rows x 16 chunks of 4 k
    const int n = idx >> 4, c4 = idx & 15;
    bf16x4 v;
#pragma unroll
    for (int j = 0; j < 4; ++j) v[j] = (__bf16)T[c4 * 4 + j][n];
    *(bf16x4*)(Wt + (size_t)(n0 + n) * K + k0 + c4 * 4) = v;
  }
}

// plain fp32 -> bf16 (vectorized)
__global__ void __launch_bounds__(256)
conv_k(const float* __restrict__ s, __bf16* __restrict__ d) {
  const size_t i = ((size_t)blockIdx.x * 256 + threadIdx.x) * 4;
  const float4 f = *(const float4*)(s + i);
  bf16x4 v; v[0] = (__bf16)f.x; v[1] = (__bf16)f.y;
  v[2] = (__bf16)f.z; v[3] = (__bf16)f.w;
  *(bf16x4*)(d + i) = v;
}

__global__ void __launch_bounds__(256)
embed_k(const int* __restrict__ ids, const float* __restrict__ E,
        const float* __restrict__ P, float* __restrict__ x) {
  const int l = blockIdx.x, tid = threadIdx.x;
  const int t = ids[l];
  const float4 e = *(const float4*)(E + (size_t)t * DMODEL + tid * 4);
  const float4 p = *(const float4*)(P + (size_t)l * DMODEL + tid * 4);
  float4 o; o.x = e.x + p.x; o.y = e.y + p.y; o.z = e.z + p.z; o.w = e.w + p.w;
  *(float4*)(x + (size_t)l * DMODEL + tid * 4) = o;
}

// rmsnorm fp32 -> bf16
__global__ void __launch_bounds__(256)
rmsnorm_k(const float* __restrict__ x, const float* __restrict__ g,
          __bf16* __restrict__ out) {
  __shared__ float red[8];
  const int l = blockIdx.x, tid = threadIdx.x;
  const float4 v = *(const float4*)(x + (size_t)l * DMODEL + tid * 4);
  float ss = v.x * v.x + v.y * v.y + v.z * v.z + v.w * v.w;
#pragma unroll
  for (int off = 1; off < 32; off <<= 1) ss += __shfl_xor(ss, off, 32);
  if ((tid & 31) == 0) red[tid >> 5] = ss;
  __syncthreads();
  float tot = 0.0f;
#pragma unroll
  for (int i = 0; i < 8; ++i) tot += red[i];
  const float rms = rsqrtf(tot * (1.0f / (float)DMODEL) + 1e-6f);
  const float4 gg = *(const float4*)(g + tid * 4);
  bf16x4 o;
  o[0] = (__bf16)(v.x * rms * gg.x); o[1] = (__bf16)(v.y * rms * gg.y);
  o[2] = (__bf16)(v.z * rms * gg.z); o[3] = (__bf16)(v.w * rms * gg.w);
  *(bf16x4*)(out + (size_t)l * DMODEL + tid * 4) = o;
}

// gv = swish(gate) * val  (fp32 in, bf16 out)
__global__ void __launch_bounds__(256)
swishmul_k(const float* __restrict__ g, const float* __restrict__ v,
           __bf16* __restrict__ out) {
  const size_t i = ((size_t)blockIdx.x * 256 + threadIdx.x) * 4;
  const float4 a = *(const float4*)(g + i);
  const float4 b = *(const float4*)(v + i);
  bf16x4 o;
  o[0] = (__bf16)(a.x / (1.0f + __expf(-a.x)) * b.x);
  o[1] = (__bf16)(a.y / (1.0f + __expf(-a.y)) * b.y);
  o[2] = (__bf16)(a.z / (1.0f + __expf(-a.z)) * b.z);
  o[3] = (__bf16)(a.w / (1.0f + __expf(-a.w)) * b.w);
  *(bf16x4*)(out + i) = o;
}

// ---------------------------------------------------------------------------
// Host orchestration
// ---------------------------------------------------------------------------
extern "C" void kernel_launch(void* const* d_in, const int* in_sizes, int n_in,
                              void* d_out, int out_size, void* d_ws, size_t ws_size,
                              hipStream_t stream) {
  (void)in_sizes; (void)n_in; (void)out_size; (void)ws_size;
  const int*   ids   = (const int*)d_in[0];
  const float* E     = (const float*)d_in[1];
  const float* P     = (const float*)d_in[2];
  const float* Wq_s  = (const float*)d_in[3];
  const float* Wk_s  = (const float*)d_in[4];
  const float* Wv_s  = (const float*)d_in[5];
  const float* Wo_s  = (const float*)d_in[6];
  const float* Wq_m  = (const float*)d_in[7];
  const float* Wdown = (const float*)d_in[8];
  const float* Wkup  = (const float*)d_in[9];
  const float* Wvup  = (const float*)d_in[10];
  const float* Wo_m  = (const float*)d_in[11];
  const float* Wgate = (const float*)d_in[12];
  const float* Wval  = (const float*)d_in[13];
  const float* Wffo  = (const float*)d_in[14];
  const float* ln1   = (const float*)d_in[15];
  const float* ln2   = (const float*)d_in[16];
  const float* oln   = (const float*)d_in[17];
  float* out = (float*)d_out;

  char* ws = (char*)d_ws;
  size_t off = 0;
  auto carve = [&](size_t bytes) { char* p = ws + off; off += (bytes + 255) & ~(size_t)255; return p; };
  float*  x    = (float*)carve((size_t)LSEQ * DMODEL * 4);
  __bf16* h1   = (__bf16*)carve((size_t)LSEQ * DMODEL * 2);
  __bf16* h2   = (__bf16*)carve((size_t)LSEQ * DMODEL * 2);
  __bf16* qbuf = (__bf16*)carve((size_t)LSEQ * DMODEL * 2);
  __bf16* kbuf = (__bf16*)carve((size_t)LSEQ * DMODEL * 2);
  __bf16* vtb  = (__bf16*)carve((size_t)DMODEL * LSEQ * 2);   // [hd_global][l]
  __bf16* ybuf = (__bf16*)carve((size_t)LSEQ * DMODEL * 2);
  __bf16* lat  = (__bf16*)carve((size_t)LSEQ * LATD * 2);
  float*  gate = (float*)carve((size_t)LSEQ * FDIM * 4);
  float*  val  = (float*)carve((size_t)LSEQ * FDIM * 4);
  __bf16* gv   = (__bf16*)carve((size_t)LSEQ * FDIM * 2);
  __bf16* wb   = (__bf16*)carve((size_t)DMODEL * FDIM * 2);   // weight scratch
  __bf16* Eb   = (__bf16*)carve((size_t)VOCAB * DMODEL * 2);  // embed bf16

  const dim3 gD (DMODEL / 64, LSEQ / 128);   // N=1024 GEMMs
  const dim3 gLat(LATD / 64, LSEQ / 128);    // N=512
  const dim3 gF (FDIM / 64, LSEQ / 128);     // N=4096
  const dim3 gV (VOCAB / 64, LSEQ / 128);    // N=32000
  const dim3 gAt(LSEQ / 64, NHEAD);
  const dim3 tDD(DMODEL / 64, DMODEL / 64);  // convtr grids (N/64, K/64)
  const dim3 tDL(LATD / 64, DMODEL / 64);
  const dim3 tLD(DMODEL / 64, LATD / 64);
  const dim3 tDF(FDIM / 64, DMODEL / 64);
  const dim3 tFD(DMODEL / 64, FDIM / 64);

  embed_k<<<LSEQ, 256, 0, stream>>>(ids, E, P, x);
  conv_k<<<(VOCAB * DMODEL / 4) / 256, 256, 0, stream>>>(E, Eb);

  const int kinds[4] = {0, 1, 0, 1};  // swa, mla, swa, mla
  int si = 0, mi = 0;
  for (int i = 0; i < 4; ++i) {
    rmsnorm_k<<<LSEQ, 256, 0, stream>>>(x, ln1 + (size_t)i * DMODEL, h1);
    rmsnorm_k<<<LSEQ, 256, 0, stream>>>(x, ln2 + (size_t)i * DMODEL, h2);
    if (kinds[i] == 0) {  // ---- sliding-window attention ----
      const size_t wo = (size_t)si * DMODEL * DMODEL;
      convtr_k<<<tDD, 256, 0, stream>>>(Wq_s + wo, wb, DMODEL, DMODEL);
      gemm_bf16<2><<<gD, 128, 0, stream>>>(h1, wb, qbuf, LSEQ, DMODEL, DMODEL);
      convtr_k<<<tDD, 256, 0, stream>>>(Wk_s + wo, wb, DMODEL, DMODEL);
      gemm_bf16<2><<<gD, 128, 0, stream>>>(h1, wb, kbuf, LSEQ, DMODEL, DMODEL);
      convtr_k<<<tDD, 256, 0, stream>>>(Wv_s + wo, wb, DMODEL, DMODEL);
      gemm_bf16<3><<<gD, 128, 0, stream>>>(h1, wb, vtb, LSEQ, DMODEL, DMODEL);
      flash_attn<1><<<gAt, 128, 0, stream>>>(qbuf, kbuf, vtb, ybuf);
      convtr_k<<<tDD, 256, 0, stream>>>(Wo_s + wo, wb, DMODEL, DMODEL);
      gemm_bf16<1><<<gD, 128, 0, stream>>>(ybuf, wb, x, LSEQ, DMODEL, DMODEL);
      ++si;
    } else {              // ---- MLA (latent) attention ----
      const size_t wq = (size_t)mi * DMODEL * DMODEL;
      const size_t wd = (size_t)mi * DMODEL * LATD;
      const size_t wu = (size_t)mi * LATD * DMODEL;
      convtr_k<<<tDD, 256, 0, stream>>>(Wq_m + wq, wb, DMODEL, DMODEL);
      gemm_bf16<2><<<gD, 128, 0, stream>>>(h1, wb, qbuf, LSEQ, DMODEL, DMODEL);
      convtr_k<<<tDL, 256, 0, stream>>>(Wdown + wd, wb, DMODEL, LATD);
      gemm_bf16<2><<<gLat, 128, 0, stream>>>(h1, wb, lat, LSEQ, LATD, DMODEL);
      convtr_k<<<tLD, 256, 0, stream>>>(Wkup + wu, wb, LATD, DMODEL);
      gemm_bf16<2><<<gD, 128, 0, stream>>>(lat, wb, kbuf, LSEQ, DMODEL, LATD);
      convtr_k<<<tLD, 256, 0, stream>>>(Wvup + wu, wb, LATD, DMODEL);
      gemm_bf16<3><<<gD, 128, 0, stream>>>(lat, wb, vtb, LSEQ, DMODEL, LATD);
      flash_attn<0><<<gAt, 128, 0, stream>>>(qbuf, kbuf, vtb, ybuf);
      convtr_k<<<tDD, 256, 0, stream>>>(Wo_m + wq, wb, DMODEL, DMODEL);
      gemm_bf16<1><<<gD, 128, 0, stream>>>(ybuf, wb, x, LSEQ, DMODEL, DMODEL);
      ++mi;
    }
    // ---- FFN ----
    const size_t wf = (size_t)i * DMODEL * FDIM;
    convtr_k<<<tDF, 256, 0, stream>>>(Wgate + wf, wb, DMODEL, FDIM);
    gemm_bf16<0><<<gF, 128, 0, stream>>>(h2, wb, gate, LSEQ, FDIM, DMODEL);
    convtr_k<<<tDF, 256, 0, stream>>>(Wval + wf, wb, DMODEL, FDIM);
    gemm_bf16<0><<<gF, 128, 0, stream>>>(h2, wb, val, LSEQ, FDIM, DMODEL);
    swishmul_k<<<(LSEQ * FDIM / 4) / 256, 256, 0, stream>>>(gate, val, gv);
    convtr_k<<<tFD, 256, 0, stream>>>(Wffo + wf, wb, FDIM, DMODEL);
    gemm_bf16<1><<<gD, 128, 0, stream>>>(gv, wb, x, LSEQ, DMODEL, FDIM);
  }

  // final norm + vocab projection (B = bf16 embed, already [V][D] = [N][K])
  rmsnorm_k<<<LSEQ, 256, 0, stream>>>(x, oln, h1);
  gemm_bf16<0><<<gV, 128, 0, stream>>>(h1, Eb, out, LSEQ, VOCAB, DMODEL);
}
